// CausalSelfAttention_26396869001352
// MI455X (gfx1250) — compile-verified, matrix operand order checked by reading, not run on hardware
//
#include <hip/hip_runtime.h>
#include <hip/hip_bf16.h>

#define N_EMBD   2048
#define N_HEAD   16
#define HEAD_DIM 128
#define BATCH    4
#define SEQ      1024

typedef __attribute__((ext_vector_type(16))) __bf16        v16bf;
typedef __attribute__((ext_vector_type(8)))  float         v8f;
typedef __attribute__((ext_vector_type(8)))  unsigned short us8v;
typedef __attribute__((ext_vector_type(4)))  unsigned short us4v;
typedef __attribute__((ext_vector_type(4)))  float         f4v;
typedef __attribute__((ext_vector_type(4)))  unsigned int   u32x4;
typedef __attribute__((ext_vector_type(8)))  int            i32x8;
typedef __attribute__((ext_vector_type(4)))  int            i32x4;

union Frag16x32 { us8v h[2]; v16bf v; };

static __device__ __forceinline__ unsigned short f2bf(float f) {
  union { float f; unsigned int u; } c; c.f = f;
  unsigned int u = c.u;
  return (unsigned short)((u + 0x7FFFu + ((u >> 16) & 1u)) >> 16);  // RNE
}
static __device__ __forceinline__ float bf2f(unsigned short s) {
  union { unsigned int u; float f; } c; c.u = ((unsigned int)s) << 16;
  return c.f;
}
static __device__ __forceinline__ v8f vzero8() {
  v8f z;
#pragma unroll
  for (int i = 0; i < 8; ++i) z[i] = 0.0f;
  return z;
}
// Load one 16x32 bf16 operand fragment: lane holds row/col (lane&15);
// lo half-wave gets K 0..7 & 16..23, hi half K 8..15 & 24..31 -> two 16B loads.
static __device__ __forceinline__ v16bf ldfrag(const unsigned short* p, int half) {
  Frag16x32 f;
  f.h[0] = *(const us8v*)(p + half * 8);
  f.h[1] = *(const us8v*)(p + 16 + half * 8);
  return f.v;
}
static __device__ __forceinline__ v8f wmma_bf16(v16bf a, v16bf b, v8f c) {
  return __builtin_amdgcn_wmma_f32_16x16x32_bf16(false, a, false, b, (short)0, c,
                                                 false, false);
}

// ---------------------------------------------------------------------------
// C[M,N] = A[M,K] * W[K,N] + bias   (row-major; bf16 WMMA, f32 accumulate)
// BM=128 BN=128 BK=32; 8 waves (2x4), wave tile 64x32 => 8 WMMA / K-step.
// ---------------------------------------------------------------------------
template <bool A_IS_BF16, bool OUT_IS_BF16>
__global__ __launch_bounds__(256, 2)
void gemm_rrr_bf16(const void* __restrict__ A_, const float* __restrict__ W,
                   const float* __restrict__ bias, void* __restrict__ C_,
                   int M, int K, int N) {
  __shared__ unsigned short Alds[128 * 40];  // [m][k], pad 32->40
  __shared__ unsigned short Blds[128 * 40];  // [n][k] (W tile transposed)

  const int tid = threadIdx.x;
  const int wv = tid >> 5, lane = tid & 31;
  const int wm = wv >> 2, wn = wv & 3;
  const int ln = lane & 15, half = lane >> 4;
  const int bm = blockIdx.y * 128;
  const int bn = blockIdx.x * 128;

  v8f acc[4][2];
#pragma unroll
  for (int i = 0; i < 4; ++i)
#pragma unroll
    for (int j = 0; j < 2; ++j) acc[i][j] = vzero8();

  const int ksteps = K >> 5;
  for (int kt = 0; kt < ksteps; ++kt) {
    const int k0 = kt << 5;
    // ---- A tile (128x32) -> LDS bf16 ----
    if constexpr (A_IS_BF16) {
      const unsigned short* A = (const unsigned short*)A_;
#pragma unroll
      for (int i = 0; i < 2; ++i) {
        int c = tid + i * 256;
        int row = c >> 2, c8 = (c & 3) << 3;
        us8v vv = *(const us8v*)(A + (size_t)(bm + row) * K + k0 + c8);
        *(us8v*)(Alds + row * 40 + c8) = vv;
      }
    } else {
      const float* A = (const float*)A_;
#pragma unroll
      for (int i = 0; i < 4; ++i) {
        int c = tid + i * 256;
        int row = c >> 3, c4 = (c & 7) << 2;
        f4v vv = *(const f4v*)(A + (size_t)(bm + row) * K + k0 + c4);
        us4v o;
        o[0] = f2bf(vv[0]); o[1] = f2bf(vv[1]);
        o[2] = f2bf(vv[2]); o[3] = f2bf(vv[3]);
        *(us4v*)(Alds + row * 40 + c4) = o;
      }
    }
    // ---- W tile (32x128) transposed -> LDS bf16 ----
#pragma unroll
    for (int i = 0; i < 4; ++i) {
      int c = tid + i * 256;
      int kr = c >> 5, c4 = (c & 31) << 2;
      f4v vv = *(const f4v*)(W + (size_t)(k0 + kr) * N + bn + c4);
#pragma unroll
      for (int j = 0; j < 4; ++j) Blds[(c4 + j) * 40 + kr] = f2bf(vv[j]);
    }
    if (kt + 1 < ksteps) {  // L2 prefetch of next W tile (global_prefetch_b8)
      __builtin_prefetch(W + (size_t)(k0 + 32 + (tid >> 5)) * N + bn + (tid & 31) * 4,
                         0, 0);
    }
    __syncthreads();

    v16bf af[4], bfv[2];
#pragma unroll
    for (int i = 0; i < 4; ++i)
      af[i] = ldfrag(Alds + (wm * 64 + i * 16 + ln) * 40, half);
#pragma unroll
    for (int j = 0; j < 2; ++j)
      bfv[j] = ldfrag(Blds + (wn * 32 + j * 16 + ln) * 40, half);
#pragma unroll
    for (int i = 0; i < 4; ++i)
#pragma unroll
      for (int j = 0; j < 2; ++j)
        acc[i][j] = wmma_bf16(af[i], bfv[j], acc[i][j]);
    __syncthreads();
  }

  // epilogue: +bias, store (C layout: lane=col, VGPR r = row r + 8*half)
#pragma unroll
  for (int i = 0; i < 4; ++i) {
#pragma unroll
    for (int j = 0; j < 2; ++j) {
      int col = bn + wn * 32 + j * 16 + ln;
      float bv = bias[col];
#pragma unroll
      for (int r = 0; r < 8; ++r) {
        int rowm = bm + wm * 64 + i * 16 + r + 8 * half;
        float v = acc[i][j][r] + bv;
        if constexpr (OUT_IS_BF16)
          ((unsigned short*)C_)[(size_t)rowm * N + col] = f2bf(v);
        else
          ((float*)C_)[(size_t)rowm * N + col] = v;
      }
    }
  }
}

// ---------------------------------------------------------------------------
// RoPE + split qkv[B,T,3,H,D] (bf16) -> qr/kr/vr[B,H,T,D] (bf16)
// ---------------------------------------------------------------------------
__global__ __launch_bounds__(64)
void rope_split(const unsigned short* __restrict__ qkv,
                unsigned short* __restrict__ qr, unsigned short* __restrict__ kr,
                unsigned short* __restrict__ vr) {
  const int gid = blockIdx.x;              // (b*SEQ + t)*N_HEAD + h
  const int h = gid & 15;
  const int t = (gid >> 4) & (SEQ - 1);
  const int b = gid >> 14;
  const int d = threadIdx.x;               // 0..63 (freq index)
  const unsigned short* row = qkv + (size_t)(b * SEQ + t) * (3 * N_EMBD);
  const int base = h * HEAD_DIM + d;

  float q1 = bf2f(row[base]);
  float q2 = bf2f(row[base + 64]);
  float k1 = bf2f(row[N_EMBD + base]);
  float k2 = bf2f(row[N_EMBD + base + 64]);

  float inv = __expf(-(float)d * (9.210340371976184f / 64.0f));  // 10000^(-d/64)
  float ang = (float)t * inv;
  float sn, cs;
  __sincosf(ang, &sn, &cs);

  size_t o = (((size_t)(b * N_HEAD + h)) * SEQ + t) * HEAD_DIM + d;
  qr[o]      = f2bf(q1 * cs - q2 * sn);
  qr[o + 64] = f2bf(q1 * sn + q2 * cs);
  kr[o]      = f2bf(k1 * cs - k2 * sn);
  kr[o + 64] = f2bf(k1 * sn + k2 * cs);
  vr[o]      = row[2 * N_EMBD + base];
  vr[o + 64] = row[2 * N_EMBD + base + 64];
}

// ---------------------------------------------------------------------------
// Flash attention forward. Grid (T/128, B*H), 256 threads = 8 waves.
// K tile fetched by the Tensor Data Mover (TDM) with hardware LDS row padding
// (128 bf16 rows padded to 136); V tile loaded+transposed by the waves.
// ---------------------------------------------------------------------------
__global__ __launch_bounds__(256, 1)
void attn_fwd(const unsigned short* __restrict__ qr,
              const unsigned short* __restrict__ kr,
              const unsigned short* __restrict__ vr,
              unsigned short* __restrict__ yb) {
  __shared__ unsigned short Klds[32 * 136];     // [key][dim], pad 128->136 (TDM pad)
  __shared__ unsigned short Vlds[128 * 40];     // [dim][key], pad 32->40
  __shared__ unsigned short Plds[8 * 16 * 40];  // per-wave P scratch

  const int tid = threadIdx.x, wv = tid >> 5, lane = tid & 31;
  const int ln = lane & 15, half = lane >> 4;
  const int bh = blockIdx.y;                    // b*16 + h
  const int b = bh >> 4, h = bh & 15;
  const int qblk = blockIdx.x;
  const int qrow = qblk * 128 + wv * 16;

  const unsigned short* qb = qr + (size_t)bh * SEQ * HEAD_DIM;
  const unsigned short* kb = kr + (size_t)bh * SEQ * HEAD_DIM;
  const unsigned short* vb = vr + (size_t)bh * SEQ * HEAD_DIM;

  // LDS byte offset of the K tile (generic LDS pointer: offset in low 32 bits)
  const unsigned kldsOff = (unsigned)(size_t)(&Klds[0]);

  v16bf qf[4];
#pragma unroll
  for (int kc = 0; kc < 4; ++kc)
    qf[kc] = ldfrag(qb + (size_t)(qrow + ln) * HEAD_DIM + kc * 32, half);

  v8f acc[8];
  float rm[8], rl[8];
#pragma unroll
  for (int d = 0; d < 8; ++d) acc[d] = vzero8();
#pragma unroll
  for (int r = 0; r < 8; ++r) { rm[r] = -3.0e38f; rl[r] = 0.0f; }

  const float scale = 0.088388347648318447f;  // 1/sqrt(128)
  const int nkt = qblk * 4 + 4;               // key tiles up to block diagonal
  for (int kt = 0; kt < nkt; ++kt) {
    // ---- K tile (32x128 bf16) via Tensor Data Mover ----
    if (wv == 0) {
      unsigned long long ga =
          (unsigned long long)(size_t)(kb + (size_t)kt * 32 * HEAD_DIM);
      // D# group0: count=1 | lds_addr | global_addr[56:0] | type=2
      u32x4 g0;
      g0[0] = 1u;
      g0[1] = kldsOff;
      g0[2] = (unsigned)(ga & 0xFFFFFFFFu);
      g0[3] = (unsigned)((ga >> 32) & 0x01FFFFFFu) | (2u << 30);
      // D# group1: data_size=2B, pad_enable, pad every 64 DWORDs (=256B row)
      // by 4 DWORDs (=8 bf16): LDS row stride 136 elements.
      i32x8 g1;
      g1[0] = (1 << 16) | (1 << 20) | (5 << 22) | (3 << 25);
      g1[1] = (int)(128u << 16);   // tensor_dim0 = 128 (elements per row)
      g1[2] = (int)(1024u << 16);  // tensor_dim1 = SEQ rows
      g1[3] = (int)(128u << 16);   // tile_dim0 = 128
      g1[4] = 32;                  // tile_dim1 = 32 rows, tile_dim2 = 0
      g1[5] = 128;                 // tensor_dim0_stride = 128 (lo 32 of 48)
      g1[6] = 0;                   // stride hi | tensor_dim1_stride lo
      g1[7] = 0;
      i32x4 gz;
      gz[0] = 0; gz[1] = 0; gz[2] = 0; gz[3] = 0;
      i32x8 gz8;
#pragma unroll
      for (int z = 0; z < 8; ++z) gz8[z] = 0;
      __builtin_amdgcn_tensor_load_to_lds(g0, g1, gz, gz, gz8, 0);
    }
    // ---- V tile (32x128) loaded + transposed by all waves ----
#pragma unroll
    for (int i = 0; i < 2; ++i) {
      int c = tid + i * 256;
      int row = c >> 4, c8 = (c & 15) << 3;
      us8v vvv = *(const us8v*)(vb + (size_t)(kt * 32 + row) * HEAD_DIM + c8);
#pragma unroll
      for (int j = 0; j < 8; ++j) Vlds[(c8 + j) * 40 + row] = vvv[j];
    }
    if (wv == 0) __builtin_amdgcn_s_wait_tensorcnt(0);  // K tile landed in LDS
    __syncthreads();

    // ---- S = Q K^T (16x32 per wave, f32 accum) ----
    v8f s[2];
#pragma unroll
    for (int j = 0; j < 2; ++j) {
      s[j] = vzero8();
      const unsigned short* kp = Klds + (j * 16 + ln) * 136;
#pragma unroll
      for (int kc = 0; kc < 4; ++kc)
        s[j] = wmma_bf16(qf[kc], ldfrag(kp + kc * 32, half), s[j]);
    }
    // ---- scale + causal mask (no divergence; arithmetic select) ----
#pragma unroll
    for (int j = 0; j < 2; ++j) {
      int kidx = kt * 32 + j * 16 + ln;
#pragma unroll
      for (int r = 0; r < 8; ++r) {
        int qidx = qrow + r + 8 * half;
        float sv = s[j][r] * scale;
        s[j][r] = (kidx > qidx) ? -3.0e38f : sv;
      }
    }
    // ---- online softmax (row reductions within 16-lane half groups) ----
#pragma unroll
    for (int r = 0; r < 8; ++r) {
      float mx = fmaxf(s[0][r], s[1][r]);
#pragma unroll
      for (int off = 8; off >= 1; off >>= 1)
        mx = fmaxf(mx, __shfl_xor(mx, off, 32));
      float mnew = fmaxf(rm[r], mx);
      float alpha = __expf(rm[r] - mnew);
      float p0 = __expf(s[0][r] - mnew);
      float p1 = __expf(s[1][r] - mnew);
      s[0][r] = p0;
      s[1][r] = p1;
      float ps = p0 + p1;
#pragma unroll
      for (int off = 8; off >= 1; off >>= 1)
        ps += __shfl_xor(ps, off, 32);
      rl[r] = rl[r] * alpha + ps;
      rm[r] = mnew;
#pragma unroll
      for (int d = 0; d < 8; ++d) acc[d][r] *= alpha;
    }
    // ---- re-layout P (C-layout -> A-fragment) via per-wave LDS scratch ----
    unsigned short* pp = Plds + wv * (16 * 40);
#pragma unroll
    for (int j = 0; j < 2; ++j)
#pragma unroll
      for (int r = 0; r < 8; ++r)
        pp[(r + 8 * half) * 40 + j * 16 + ln] = f2bf(s[j][r]);
    v16bf pf = ldfrag(pp + ln * 40, half);
    // ---- O += P V ----
#pragma unroll
    for (int d = 0; d < 8; ++d)
      acc[d] = wmma_bf16(pf, ldfrag(Vlds + (d * 16 + ln) * 40, half), acc[d]);
    __syncthreads();
  }

  // ---- normalize, store y[b, t, h*128 + col] as bf16 ----
#pragma unroll
  for (int d = 0; d < 8; ++d) {
    int col = h * HEAD_DIM + d * 16 + ln;
#pragma unroll
    for (int r = 0; r < 8; ++r) {
      int qidx = qrow + r + 8 * half;
      float v = acc[d][r] / rl[r];
      yb[((size_t)(b * SEQ + qidx)) * N_EMBD + col] = f2bf(v);
    }
  }
}

// ---------------------------------------------------------------------------
extern "C" void kernel_launch(void* const* d_in, const int* in_sizes, int n_in,
                              void* d_out, int out_size, void* d_ws, size_t ws_size,
                              hipStream_t stream) {
  (void)in_sizes; (void)n_in; (void)out_size; (void)ws_size;
  const float* x     = (const float*)d_in[0];
  const float* Wqkv  = (const float*)d_in[1];
  const float* bqkv  = (const float*)d_in[2];
  const float* Wproj = (const float*)d_in[3];
  const float* bproj = (const float*)d_in[4];

  const size_t M = (size_t)BATCH * SEQ;               // 4096 rows
  unsigned short* qkv = (unsigned short*)d_ws;        // [4096, 6144] bf16
  unsigned short* qr  = qkv + M * (3 * N_EMBD);
  unsigned short* kr  = qr + (size_t)BATCH * N_HEAD * SEQ * HEAD_DIM;
  unsigned short* vr  = kr + (size_t)BATCH * N_HEAD * SEQ * HEAD_DIM;
  unsigned short* yb  = qkv;  // reuse: qkv buffer is dead after rope_split

  gemm_rrr_bf16<false, true><<<dim3(3 * N_EMBD / 128, M / 128), 256, 0, stream>>>(
      x, Wqkv, bqkv, qkv, (int)M, N_EMBD, 3 * N_EMBD);
  rope_split<<<BATCH * SEQ * N_HEAD, 64, 0, stream>>>(qkv, qr, kr, vr);
  attn_fwd<<<dim3(SEQ / 128, BATCH * N_HEAD), 256, 0, stream>>>(qr, kr, vr, yb);
  gemm_rrr_bf16<true, false><<<dim3(N_EMBD / 128, M / 128), 256, 0, stream>>>(
      yb, Wproj, bproj, (float*)d_out, (int)M, N_EMBD, N_EMBD);
}
